// MultiHeadAttention_57501022159163
// MI455X (gfx1250) — matrix operand order repaired and checked by reading, not compiled, not run on hardware
//
#include <hip/hip_runtime.h>
#include <hip/hip_bf16.h>
#include <stdint.h>

// ---------------------------------------------------------------------------
// MHA for MI455X (gfx1250): bf16 WMMA everywhere, f32 accumulate.
// GEMM: per-block async-to-LDS staging of the A panel, 16x64 register tiles,
// and software-pipelined (double-buffered) fragment loads.
// ---------------------------------------------------------------------------

typedef __attribute__((ext_vector_type(16))) __bf16 v16bf;
typedef __attribute__((ext_vector_type(8)))  __bf16 v8bf;
typedef __attribute__((ext_vector_type(8)))  float  v8f;

#define HH 16
#define DD 64
#define EE 1024
#define NN 2048
#define BB 2
#define MTOT (BB * NN)   // 4096 tokens

// ---- fp32 -> bf16 elementwise conversion -----------------------------------
__global__ void __launch_bounds__(256) cvt_f32_bf16(const float* __restrict__ in,
                                                    __bf16* __restrict__ out, int n) {
  int i = blockIdx.x * blockDim.x + threadIdx.x;
  int stride = gridDim.x * blockDim.x;
  for (; i < n; i += stride) out[i] = (__bf16)in[i];
}

// ---- build A fragment (16x32 bf16) from row-major storage ------------------
// Contiguous K order (matches B): lanes 0-15 hold row M=lane, elems = K{0..15};
// lanes 16-31 hold row M=lane-16, elems = K{16..31}.
__device__ __forceinline__ v16bf load_afrag(const __bf16* base, int row_stride, int lane) {
  const int row = lane & 15;
  const int kb  = (lane >> 4) << 4;  // 0 or 16
  const __bf16* p = base + (size_t)row * row_stride + kb;
  v8bf lo = *(const v8bf*)(p);
  v8bf hi = *(const v8bf*)(p + 8);
  v16bf a;
#pragma unroll
  for (int i = 0; i < 8; ++i) { a[i] = lo[i]; a[i + 8] = hi[i]; }
  return a;
}

#define WMMA_BF16(A_, B_, C_) \
  __builtin_amdgcn_wmma_f32_16x16x32_bf16(false, (A_), false, (B_), (short)0, (C_), false, false)

// ---- WMMA GEMM: C[M,N] = A[M,K] * W[N,K]^T + bias ---------------------------
// One block = 8 waves sharing one 16-row A panel (async-staged into LDS once);
// each wave computes a 16x64 tile. Fragments are double-buffered one K-step
// ahead so the 4 WMMAs of step i cover the load latency of step i+1.
// mode 0: f32 row-major out[m*N + n]            (final projection)
// mode 1: bf16 head-split  out[b][h][npos][d]   (q, k)
// mode 2: bf16 head-split transposed out[b][h][d][npos]  (v^T)
__global__ void __launch_bounds__(256) mha_gemm_wmma(
    const __bf16* __restrict__ A, const __bf16* __restrict__ W,
    const float* __restrict__ bias, void* __restrict__ out,
    int M, int K, int Nout, int mode) {
  __shared__ __align__(16) __bf16 As[16 * EE];   // 32 KB A panel [16][K]

  const int tid    = threadIdx.x;
  const int lane   = tid & 31;
  const int wv     = tid >> 5;          // 0..7
  const int nblk64 = Nout >> 6;
  const int wid    = blockIdx.x * 8 + wv;
  const int m0     = (wid / nblk64) << 4;   // uniform across the block
  const int n0     = (wid % nblk64) << 6;
  if (m0 >= M) return;

  // ---- async-stage A[m0:m0+16, 0:K] into LDS (ASYNCcnt path) ----
  {
    const __bf16*  gsrc  = A + (size_t)m0 * K;
    const uint32_t lbase = (uint32_t)(uintptr_t)(&As[0]);  // low 32b of generic = LDS offset
#pragma unroll
    for (int i = 0; i < 8; ++i) {
      const int      idx  = tid + i * 256;            // 16-byte chunk index (2048 total)
      const uint32_t loff = lbase + (uint32_t)idx * 16u;
      const uint64_t gadr = (uint64_t)(uintptr_t)(gsrc + idx * 8);
      asm volatile("global_load_async_to_lds_b128 %0, %1, off"
                   :: "v"(loff), "v"(gadr) : "memory");
    }
    asm volatile("s_wait_asynccnt 0x0" ::: "memory");
    __syncthreads();
  }

  const int col   = lane & 15;
  const int koffB = (lane >> 4) << 4;   // B fragment: +0 / +16 within column
  const __bf16* Wp0 = W + (size_t)(n0 +      col) * K + koffB;
  const __bf16* Wp1 = W + (size_t)(n0 + 16 + col) * K + koffB;
  const __bf16* Wp2 = W + (size_t)(n0 + 32 + col) * K + koffB;
  const __bf16* Wp3 = W + (size_t)(n0 + 48 + col) * K + koffB;

  v8f acc0 = {}, acc1 = {}, acc2 = {}, acc3 = {};

  // prologue: fragments for k0 = 0
  v16bf a_c  = load_afrag(&As[0], K, lane);
  v16bf b0c = *(const v16bf*)(Wp0);
  v16bf b1c = *(const v16bf*)(Wp1);
  v16bf b2c = *(const v16bf*)(Wp2);
  v16bf b3c = *(const v16bf*)(Wp3);

  // steady state: issue k0+32 loads, then run k0's WMMAs over them
  for (int k0 = 0; k0 + 32 < K; k0 += 32) {
    const int kn = k0 + 32;
    __builtin_prefetch(Wp0 + kn + 256, 0, 1);   // global_prefetch_b8
    __builtin_prefetch(Wp1 + kn + 256, 0, 1);
    __builtin_prefetch(Wp2 + kn + 256, 0, 1);
    __builtin_prefetch(Wp3 + kn + 256, 0, 1);
    const v16bf a_n  = load_afrag(&As[0] + kn, K, lane);
    const v16bf b0n = *(const v16bf*)(Wp0 + kn);
    const v16bf b1n = *(const v16bf*)(Wp1 + kn);
    const v16bf b2n = *(const v16bf*)(Wp2 + kn);
    const v16bf b3n = *(const v16bf*)(Wp3 + kn);

    acc0 = WMMA_BF16(a_c, b0c, acc0);
    acc1 = WMMA_BF16(a_c, b1c, acc1);
    acc2 = WMMA_BF16(a_c, b2c, acc2);
    acc3 = WMMA_BF16(a_c, b3c, acc3);

    a_c = a_n; b0c = b0n; b1c = b1n; b2c = b2n; b3c = b3n;
  }
  // epilogue: last K-step
  acc0 = WMMA_BF16(a_c, b0c, acc0);
  acc1 = WMMA_BF16(a_c, b1c, acc1);
  acc2 = WMMA_BF16(a_c, b2c, acc2);
  acc3 = WMMA_BF16(a_c, b3c, acc3);

  const int rbase = (lane >> 4) << 3;   // C-layout: upper lanes hold rows 8..15
  v8f accs[4] = {acc0, acc1, acc2, acc3};
#pragma unroll
  for (int t = 0; t < 4; ++t) {
    const int nn = n0 + t * 16 + col;
    const float bv = bias[nn];
    if (mode == 0) {
      float* o = (float*)out;
#pragma unroll
      for (int r = 0; r < 8; ++r) {
        const int m = m0 + r + rbase;
        o[(size_t)m * Nout + nn] = accs[t][r] + bv;
      }
    } else {
      __bf16* o = (__bf16*)out;
      const int hh = nn >> 6, dd = nn & 63;
#pragma unroll
      for (int r = 0; r < 8; ++r) {
        const int m  = m0 + r + rbase;
        const int bb = m >> 11, np = m & (NN - 1);
        const float v = accs[t][r] + bv;
        const size_t idx = (mode == 1)
          ? ((((size_t)bb * HH + hh) * NN + np) * DD + dd)
          : ((((size_t)bb * HH + hh) * DD + dd) * NN + np);
        o[idx] = (__bf16)v;
      }
    }
  }
}

// ---- flash attention: 4 waves/block, one (b, h, 16-query tile) per wave -----
__global__ void __launch_bounds__(128) mha_flash_wmma(
    const __bf16* __restrict__ q,    // [B,H,N,D]
    const __bf16* __restrict__ k,    // [B,H,N,D]
    const __bf16* __restrict__ vt,   // [B,H,D,N]
    const uint8_t* __restrict__ mask,// [B,N,N] bool
    __bf16* __restrict__ xout) {     // [B,N,E]
  __shared__ __align__(32) __bf16 Pl[4][16][32];

  const int lane = threadIdx.x & 31;
  const int wv   = threadIdx.x >> 5;
  int bid = blockIdx.x * 4 + wv;
  const int qt = bid & 127; bid >>= 7;
  const int h  = bid & 15;  bid >>= 4;
  const int b  = bid;

  const size_t head = (size_t)b * HH + h;
  const __bf16* qp = q  + (head * NN + (size_t)qt * 16) * DD;
  const __bf16* kp = k  + head * NN * DD;
  const __bf16* vp = vt + head * DD * NN;
  const uint8_t* mp = mask + ((size_t)b * NN + (size_t)qt * 16) * NN;

  // Q fragments for d-chunks 0..31 and 32..63, held across the key loop.
  const v16bf aq0 = load_afrag(qp,      DD, lane);
  const v16bf aq1 = load_afrag(qp + 32, DD, lane);

  v8f acc0 = {}, acc1 = {}, acc2 = {}, acc3 = {};
  float mrun[8], lrun[8];
#pragma unroll
  for (int r = 0; r < 8; ++r) { mrun[r] = -1e30f; lrun[r] = 0.f; }

  const int col   = lane & 15;
  const int rbase = (lane >> 4) << 3;
  const int koffB = (lane >> 4) << 4;

  for (int kc = 0; kc < NN; kc += 32) {
    // ---- scores: two 16-key tiles, K-loop over D in chunks of 32 ----
    v8f s0 = {}, s1 = {};
    const __bf16* kb0 = kp + (size_t)(kc + col) * DD + koffB;
    const __bf16* kb1 = kp + (size_t)(kc + 16 + col) * DD + koffB;
    s0 = WMMA_BF16(aq0, *(const v16bf*)(kb0),      s0);
    s0 = WMMA_BF16(aq1, *(const v16bf*)(kb0 + 32), s0);
    s1 = WMMA_BF16(aq0, *(const v16bf*)(kb1),      s1);
    s1 = WMMA_BF16(aq1, *(const v16bf*)(kb1 + 32), s1);

    // ---- scale + mask (C-layout: lane = key column, VGPR = query row) ----
#pragma unroll
    for (int r = 0; r < 8; ++r) {
      const int row = r + rbase;
      float v0 = s0[r] * 0.125f;   // 1/sqrt(D), D=64
      float v1 = s1[r] * 0.125f;
      if (!mp[(size_t)row * NN + kc + col])      v0 = -1e30f;
      if (!mp[(size_t)row * NN + kc + 16 + col]) v1 = -1e30f;
      s0[r] = v0; s1[r] = v1;
    }

    // ---- online softmax: xor-butterfly over the 16-lane half wave ----
#pragma unroll
    for (int r = 0; r < 8; ++r) {
      float mv = fmaxf(s0[r], s1[r]);
#pragma unroll
      for (int off = 1; off < 16; off <<= 1) mv = fmaxf(mv, __shfl_xor(mv, off, 32));
      const float mnew  = fmaxf(mrun[r], mv);
      const float scale = __expf(mrun[r] - mnew);
      const float p0 = __expf(s0[r] - mnew);
      const float p1 = __expf(s1[r] - mnew);
      float rs = p0 + p1;
#pragma unroll
      for (int off = 1; off < 16; off <<= 1) rs += __shfl_xor(rs, off, 32);
      lrun[r] = lrun[r] * scale + rs;
      mrun[r] = mnew;
      acc0[r] *= scale; acc1[r] *= scale; acc2[r] *= scale; acc3[r] *= scale;
      // stage P (C-layout -> row-major LDS) for re-read as an A fragment
      Pl[wv][r + rbase][col]      = (__bf16)p0;
      Pl[wv][r + rbase][16 + col] = (__bf16)p1;
    }
    __syncthreads();
    const v16bf ap = load_afrag(&Pl[wv][0][0], 32, lane);
    __syncthreads();

    // ---- P (16x32) x V-chunk (32x64): B fragments from transposed V ----
    const __bf16* vb = vp + kc + koffB;
    acc0 = WMMA_BF16(ap, *(const v16bf*)(vb + (size_t)(col)      * NN), acc0);
    acc1 = WMMA_BF16(ap, *(const v16bf*)(vb + (size_t)(16 + col) * NN), acc1);
    acc2 = WMMA_BF16(ap, *(const v16bf*)(vb + (size_t)(32 + col) * NN), acc2);
    acc3 = WMMA_BF16(ap, *(const v16bf*)(vb + (size_t)(48 + col) * NN), acc3);
  }

  // ---- epilogue: normalize and write x[b, npos, h*64 + d] as bf16 ----
  __bf16* xo = xout + ((size_t)b * NN + (size_t)qt * 16) * EE + (size_t)h * DD;
#pragma unroll
  for (int r = 0; r < 8; ++r) {
    const float inv = 1.0f / lrun[r];
    const int row = r + rbase;
    xo[(size_t)row * EE +      col] = (__bf16)(acc0[r] * inv);
    xo[(size_t)row * EE + 16 + col] = (__bf16)(acc1[r] * inv);
    xo[(size_t)row * EE + 32 + col] = (__bf16)(acc2[r] * inv);
    xo[(size_t)row * EE + 48 + col] = (__bf16)(acc3[r] * inv);
  }
}

// ---------------------------------------------------------------------------
extern "C" void kernel_launch(void* const* d_in, const int* in_sizes, int n_in,
                              void* d_out, int out_size, void* d_ws, size_t ws_size,
                              hipStream_t stream) {
  const float*   Q    = (const float*)d_in[0];
  const float*   Kin  = (const float*)d_in[1];
  const float*   Vin  = (const float*)d_in[2];
  const uint8_t* mask = (const uint8_t*)d_in[3];
  const float *Wq = (const float*)d_in[4],  *bq = (const float*)d_in[5];
  const float *Wk = (const float*)d_in[6],  *bk = (const float*)d_in[7];
  const float *Wv = (const float*)d_in[8],  *bv = (const float*)d_in[9];
  const float *Wo = (const float*)d_in[10], *bo = (const float*)d_in[11];

  char* ws = (char*)d_ws;
  const size_t MB = 1ull << 20;
  __bf16* Xq  = (__bf16*)(ws +  0 * MB);   // [4096,1024] bf16  (8 MB each)
  __bf16* Xk  = (__bf16*)(ws +  8 * MB);
  __bf16* Xv  = (__bf16*)(ws + 16 * MB);
  __bf16* Wqb = (__bf16*)(ws + 24 * MB);   // [1024,1024] bf16  (2 MB each)
  __bf16* Wkb = (__bf16*)(ws + 26 * MB);
  __bf16* Wvb = (__bf16*)(ws + 28 * MB);
  __bf16* Wob = (__bf16*)(ws + 30 * MB);
  __bf16* qh  = (__bf16*)(ws + 32 * MB);   // [B,H,N,D] bf16
  __bf16* kh  = (__bf16*)(ws + 40 * MB);   // [B,H,N,D] bf16
  __bf16* vth = (__bf16*)(ws + 48 * MB);   // [B,H,D,N] bf16 (transposed V)
  __bf16* xb  = (__bf16*)(ws + 56 * MB);   // [B,N,E]   bf16 attention output

  const int NE = MTOT * EE;   // 4M elems
  const int WE = EE * EE;     // 1M elems
  cvt_f32_bf16<<<2048, 256, 0, stream>>>(Q,   Xq,  NE);
  cvt_f32_bf16<<<2048, 256, 0, stream>>>(Kin, Xk,  NE);
  cvt_f32_bf16<<<2048, 256, 0, stream>>>(Vin, Xv,  NE);
  cvt_f32_bf16<<<1024, 256, 0, stream>>>(Wq,  Wqb, WE);
  cvt_f32_bf16<<<1024, 256, 0, stream>>>(Wk,  Wkb, WE);
  cvt_f32_bf16<<<1024, 256, 0, stream>>>(Wv,  Wvb, WE);
  cvt_f32_bf16<<<1024, 256, 0, stream>>>(Wo,  Wob, WE);

  const int waves = (MTOT / 16) * (EE / 64);   // 4096 wave-tiles (16x64 each)
  dim3 gg(waves / 8), gb(256);                 // 8 waves / block (wave32)
  mha_gemm_wmma<<<gg, gb, 0, stream>>>(Xq, Wqb, bq, qh,  MTOT, EE, EE, 1);
  mha_gemm_wmma<<<gg, gb, 0, stream>>>(Xk, Wkb, bk, kh,  MTOT, EE, EE, 1);
  mha_gemm_wmma<<<gg, gb, 0, stream>>>(Xv, Wvb, bv, vth, MTOT, EE, EE, 2);

  mha_flash_wmma<<<dim3(BB * HH * (NN / 16) / 4), dim3(128), 0, stream>>>(qh, kh, vth, mask, xb);

  mha_gemm_wmma<<<gg, gb, 0, stream>>>(xb, Wob, bo, d_out, MTOT, EE, EE, 0);
}
